// TransformerModule_82265803587776
// MI455X (gfx1250) — matrix operand order repaired and checked by reading, not compiled, not run on hardware
//
#include <hip/hip_runtime.h>
#include <math.h>

// Problem constants (match reference)
#define Dm   256
#define Hh   8
#define DKh  32
#define Ll   6
#define DFFf 1024
#define Tt   256
#define Bb   128
#define Mrows (Tt * Bb) // 32768

typedef __attribute__((ext_vector_type(16))) __bf16 v16bf;
typedef __attribute__((ext_vector_type(4)))  __bf16 v4bf;
typedef __attribute__((ext_vector_type(8)))  float  v8f;

__device__ __forceinline__ float sigmoidf_(float x) { return 1.0f / (1.0f + __expf(-x)); }

__device__ __forceinline__ v4bf pack4_bf16(float4 v) {
    v4bf p;
    p[0] = (__bf16)v.x; p[1] = (__bf16)v.y; p[2] = (__bf16)v.z; p[3] = (__bf16)v.w;
    return p;
}

// ---------------------------------------------------------------------------
// GEMM: C[M x N] = A[M x K] * W[N x K]^T (+bias)(+relu), bf16 WMMA, f32 accum.
// A may be split at column K1 (multiple of 32) across two row-major sources
// A0 (lda0) / A1 (lda1) to implement concat(x, y) without materializing it.
// Block = 256 threads (8 waves), tile = 128(M) x 64(N), k-step = 32.
// Software-pipelined: slab k+1 is fetched into registers while slab k's
// WMMAs execute, hiding global-load latency behind the matrix pipe.
// ---------------------------------------------------------------------------
__global__ __launch_bounds__(256)
void gemm_bf16_wmma(const float* __restrict__ A0, int lda0,
                    const float* __restrict__ A1, int lda1, int K1,
                    const float* __restrict__ W,
                    const float* __restrict__ bias,
                    float* __restrict__ C,
                    int N, int K, int relu)
{
    __shared__ __bf16 As[128 * 32];   // 8 KB
    __shared__ __bf16 Bs[64 * 32];    // 4 KB

    const int m0   = blockIdx.y << 7;
    const int n0   = blockIdx.x << 6;
    const int tid  = threadIdx.x;
    const int lane = tid & 31;
    const int w    = tid >> 5;

    v8f acc[4];
#pragma unroll
    for (int i = 0; i < 4; ++i) acc[i] = (v8f){};

    float4 ra[4];   // staged A slab (128x32: 1024 float4 / 256 threads)
    float4 rb[2];   // staged W slab (64x32:   512 float4 / 256 threads)

    // Fetch one k-slab into registers (uniform A0/A1 selection per slab:
    // K1 is a multiple of 32, so a slab never straddles the concat boundary).
    auto fetch = [&](int kk) {
        const float* Ap; int lda, ka;
        if (kk < K1) { Ap = A0; lda = lda0; ka = kk; }
        else         { Ap = A1; lda = lda1; ka = kk - K1; }
#pragma unroll
        for (int u = 0; u < 4; ++u) {
            int q  = tid + (u << 8);
            int r  = q >> 3;            // 0..127
            int c4 = (q & 7) << 2;      // 0,4,..,28
            ra[u] = *(const float4*)&Ap[(size_t)(m0 + r) * lda + ka + c4];
        }
#pragma unroll
        for (int u = 0; u < 2; ++u) {
            int q  = tid + (u << 8);
            int r  = q >> 3;            // 0..63
            int c4 = (q & 7) << 2;
            rb[u] = *(const float4*)&W[(size_t)(n0 + r) * K + kk + c4];
        }
    };

    fetch(0);
    for (int kk = 0; kk < K; kk += 32) {
        // Commit staged registers to LDS (f32 -> packed bf16x4).
#pragma unroll
        for (int u = 0; u < 4; ++u) {
            int q = tid + (u << 8);
            *(v4bf*)&As[(q >> 3) * 32 + ((q & 7) << 2)] = pack4_bf16(ra[u]);
        }
#pragma unroll
        for (int u = 0; u < 2; ++u) {
            int q = tid + (u << 8);
            *(v4bf*)&Bs[(q >> 3) * 32 + ((q & 7) << 2)] = pack4_bf16(rb[u]);
        }
        __syncthreads();

        // Kick off next slab's global loads; they retire during the WMMAs.
        if (kk + 32 < K) fetch(kk + 32);

        // A fragment (16x32 bf16, wave32 layout): lanes 0-15 rows, K 0-7/16-23;
        // lanes 16-31 same rows, K 8-15/24-31.
        v16bf a;
        {
            int r  = (w << 4) | (lane & 15);
            int kb = (lane & 16) ? 8 : 0;
#pragma unroll
            for (int j = 0; j < 16; ++j) {
                int k = kb + ((j < 8) ? j : j + 8);
                a[j] = As[r * 32 + k];
            }
        }
        // Load ALL B fragments first (one ds clause + single wait), then issue
        // four independent WMMAs back-to-back into the matrix pipe.
        v16bf bf[4];
#pragma unroll
        for (int nt = 0; nt < 4; ++nt) {
            int c   = (nt << 4) | (lane & 15);
            int kb2 = (lane & 16) ? 8 : 0;
#pragma unroll
            for (int j = 0; j < 16; ++j) bf[nt][j] = Bs[c * 32 + kb2 + ((j < 8) ? j : j + 8)];
        }
#pragma unroll
        for (int nt = 0; nt < 4; ++nt)
            acc[nt] = __builtin_amdgcn_wmma_f32_16x16x32_bf16(
                false, a, false, bf[nt], (short)0, acc[nt], false, false);
        __syncthreads();
    }

    // C layout: vgpr e -> row e (+8 for upper lane half), lane%16 -> column.
#pragma unroll
    for (int nt = 0; nt < 4; ++nt) {
#pragma unroll
        for (int e = 0; e < 8; ++e) {
            int row = (w << 4) + e + ((lane & 16) ? 8 : 0);
            int gn  = n0 + ((nt << 4) | (lane & 15));
            float v = acc[nt][e];
            if (bias) v += bias[gn];
            if (relu) v = fmaxf(v, 0.0f);
            C[(size_t)(m0 + row) * N + gn] = v;
        }
    }
}

// ---------------------------------------------------------------------------
// Flash-style causal attention. One block per (batch, head); 8 waves; each
// wave owns two 16-row strips. K/V staged in LDS as bf16; P tile bounced
// through LDS to re-layout C-fragment -> A-fragment for the P*V WMMA.
// ---------------------------------------------------------------------------
__global__ __launch_bounds__(256)
void attn_wmma(const float* __restrict__ qkv, float* __restrict__ o)
{
    __shared__ __bf16 Kl[Tt * DKh];       // 16 KB
    __shared__ __bf16 Vl[Tt * DKh];       // 16 KB
    __shared__ __bf16 Pl[8][16 * 32];     // 8 KB (1 KB per wave)

    const int b    = blockIdx.x / Hh;
    const int hh   = blockIdx.x % Hh;
    const int tid  = threadIdx.x;
    const int lane = tid & 31;
    const int w    = tid >> 5;
    const float scale = 0.17677669529663687f; // 1/sqrt(32)

    // Vectorized K/V staging: float4 global loads, bf16x4 LDS stores.
#pragma unroll
    for (int q = tid; q < (Tt * DKh) / 4; q += 256) {
        int i = q << 2;                 // element index, d-aligned to 4
        int s = i >> 5, d = i & 31;
        size_t base = ((size_t)(s * Bb + b)) * (3 * Dm) + hh * DKh + d;
        float4 kv = *(const float4*)&qkv[base + Dm];
        float4 vv = *(const float4*)&qkv[base + 2 * Dm];
        *(v4bf*)&Kl[i] = pack4_bf16(kv);
        *(v4bf*)&Vl[i] = pack4_bf16(vv);
    }
    __syncthreads();

    for (int rs = 0; rs < 2; ++rs) {
        const int r0 = w * 32 + rs * 16;

        // Q fragment from global: 4 float4 loads per lane (k = kb..kb+7, kb+16..kb+23).
        v16bf qf;
        {
            int t  = r0 + (lane & 15);
            int kb = (lane & 16) ? 8 : 0;
            size_t base = ((size_t)(t * Bb + b)) * (3 * Dm) + hh * DKh + kb;
            float4 q0 = *(const float4*)&qkv[base];
            float4 q1 = *(const float4*)&qkv[base + 4];
            float4 q2 = *(const float4*)&qkv[base + 16];
            float4 q3 = *(const float4*)&qkv[base + 20];
            qf[0]  = (__bf16)q0.x; qf[1]  = (__bf16)q0.y; qf[2]  = (__bf16)q0.z; qf[3]  = (__bf16)q0.w;
            qf[4]  = (__bf16)q1.x; qf[5]  = (__bf16)q1.y; qf[6]  = (__bf16)q1.z; qf[7]  = (__bf16)q1.w;
            qf[8]  = (__bf16)q2.x; qf[9]  = (__bf16)q2.y; qf[10] = (__bf16)q2.z; qf[11] = (__bf16)q2.w;
            qf[12] = (__bf16)q3.x; qf[13] = (__bf16)q3.y; qf[14] = (__bf16)q3.z; qf[15] = (__bf16)q3.w;
        }

        v8f acc0 = (v8f){}, acc1 = (v8f){};
        float mrow[8], srow[8];
#pragma unroll
        for (int e = 0; e < 8; ++e) { mrow[e] = -1e30f; srow[e] = 0.0f; }

        const int npair = ((r0 + 15) >> 5) + 1;   // causal: only s <= row
        for (int jp = 0; jp < npair; ++jp) {
            const int s0 = jp * 32;

            // Load both K fragments, then two back-to-back QK^T WMMAs.
            v16bf kf[2];
#pragma unroll
            for (int t2 = 0; t2 < 2; ++t2) {
                int n   = s0 + (t2 << 4) + (lane & 15);
                int kb2 = (lane & 16) ? 8 : 0;
#pragma unroll
                for (int j = 0; j < 16; ++j) kf[t2][j] = Kl[n * 32 + kb2 + ((j < 8) ? j : j + 8)];
            }
            v8f z0 = (v8f){}, z1 = (v8f){};
            v8f S0 = __builtin_amdgcn_wmma_f32_16x16x32_bf16(
                false, qf, false, kf[0], (short)0, z0, false, false);
            v8f S1 = __builtin_amdgcn_wmma_f32_16x16x32_bf16(
                false, qf, false, kf[1], (short)0, z1, false, false);

            float st[2][8];
#pragma unroll
            for (int e = 0; e < 8; ++e) {
                int row = r0 + e + ((lane & 16) ? 8 : 0);
                int col = s0 + (lane & 15);
                st[0][e] = (col <= row)      ? S0[e] * scale : -1e30f;
                st[1][e] = (col + 16 <= row) ? S1[e] * scale : -1e30f;
            }

            // Online softmax update (row reductions across 16-lane halves).
#pragma unroll
            for (int e = 0; e < 8; ++e) {
                float tm = fmaxf(st[0][e], st[1][e]);
#pragma unroll
                for (int off = 1; off < 16; off <<= 1)
                    tm = fmaxf(tm, __shfl_xor(tm, off, 32));
                float nm    = fmaxf(mrow[e], tm);
                float alpha = __expf(mrow[e] - nm);
                float p0    = __expf(st[0][e] - nm);
                float p1    = __expf(st[1][e] - nm);
                float rsum  = p0 + p1;
#pragma unroll
                for (int off = 1; off < 16; off <<= 1)
                    rsum += __shfl_xor(rsum, off, 32);
                srow[e] = srow[e] * alpha + rsum;
                mrow[e] = nm;
                acc0[e] = acc0[e] * alpha;
                acc1[e] = acc1[e] * alpha;
                int rl = e + ((lane & 16) ? 8 : 0);
                int cl = lane & 15;
                Pl[w][rl * 32 + cl]      = (__bf16)p0;
                Pl[w][rl * 32 + 16 + cl] = (__bf16)p1;
            }
            // Wave-internal LDS RAW: make P writes visible before fragment reads.
            asm volatile("s_wait_dscnt 0" ::: "memory");

            // P fragment + both V fragments, then two back-to-back PV WMMAs.
            v16bf pf;
            {
                int rl = lane & 15;
                int kb = (lane & 16) ? 8 : 0;
#pragma unroll
                for (int j = 0; j < 16; ++j) {
                    int k = kb + ((j < 8) ? j : j + 8);
                    pf[j] = Pl[w][rl * 32 + k];
                }
            }
            v16bf vf[2];
#pragma unroll
            for (int dt = 0; dt < 2; ++dt) {
                int d   = (dt << 4) + (lane & 15);
                int kb2 = (lane & 16) ? 8 : 0;
#pragma unroll
                for (int j = 0; j < 16; ++j) vf[dt][j] = Vl[(s0 + kb2 + ((j < 8) ? j : j + 8)) * 32 + d];
            }
            acc0 = __builtin_amdgcn_wmma_f32_16x16x32_bf16(
                false, pf, false, vf[0], (short)0, acc0, false, false);
            acc1 = __builtin_amdgcn_wmma_f32_16x16x32_bf16(
                false, pf, false, vf[1], (short)0, acc1, false, false);
        }

#pragma unroll
        for (int e = 0; e < 8; ++e) {
            float inv = 1.0f / srow[e];
            int t  = r0 + e + ((lane & 16) ? 8 : 0);
            int cl = lane & 15;
            size_t base = ((size_t)(t * Bb + b)) * Dm + hh * DKh;
            o[base + cl]      = acc0[e] * inv;
            o[base + 16 + cl] = acc1[e] * inv;
        }
    }
}

// ---------------------------------------------------------------------------
// LayerNorm over D=256; one block per row.
// ---------------------------------------------------------------------------
__global__ __launch_bounds__(256)
void layernorm_k(const float* __restrict__ x, const float* __restrict__ wt,
                 const float* __restrict__ bs, float* __restrict__ y)
{
    __shared__ float red[256];
    const int m = blockIdx.x;
    const int d = threadIdx.x;
    float v = x[(size_t)m * Dm + d];
    red[d] = v;
    __syncthreads();
#pragma unroll
    for (int s = 128; s > 0; s >>= 1) { if (d < s) red[d] += red[d + s]; __syncthreads(); }
    float mean = red[0] * (1.0f / Dm);
    __syncthreads();
    float c = v - mean;
    red[d] = c * c;
    __syncthreads();
#pragma unroll
    for (int s = 128; s > 0; s >>= 1) { if (d < s) red[d] += red[d + s]; __syncthreads(); }
    float var = red[0] * (1.0f / Dm);
    y[(size_t)m * Dm + d] = c * rsqrtf(var + 1e-5f) * wt[d] + bs[d];
}

// Sinusoidal positional encoding, computed on the fly (scaled by 0.1).
__global__ void add_pe_k(float* __restrict__ h)
{
    int idx = blockIdx.x * blockDim.x + threadIdx.x; // < M*D
    int d  = idx & (Dm - 1);
    int t  = idx >> 15;                              // / (B*D) = 32768
    int d2 = d & ~1;
    float div = __expf((float)d2 * (-9.210340371976184f / (float)Dm));
    float ang = (float)t * div;
    float pe  = ((d & 1) ? __cosf(ang) : __sinf(ang)) * 0.1f;
    h[idx] += pe;
}

// Gated residual update: h = (1-z)*h + z*tanh(g2*r), r=sig(g0), z=sig(g1-bg).
__global__ void gate_k(float* __restrict__ h, const float* __restrict__ g,
                       const float* __restrict__ bg)
{
    int idx = blockIdx.x * blockDim.x + threadIdx.x;
    int d = idx & (Dm - 1);
    size_t m = (size_t)idx >> 8;
    const float* gr = g + m * (3 * Dm);
    float r  = sigmoidf_(gr[d]);
    float z  = sigmoidf_(gr[Dm + d] - bg[d]);
    float hc = tanhf(gr[2 * Dm + d] * r);
    float xv = h[idx];
    h[idx] = (1.0f - z) * xv + z * hc;
}

// ---------------------------------------------------------------------------
extern "C" void kernel_launch(void* const* d_in, const int* in_sizes, int n_in,
                              void* d_out, int out_size, void* d_ws, size_t ws_size,
                              hipStream_t stream)
{
    const float* x    = (const float*)d_in[0];
    const float* in_w = (const float*)d_in[1];
    const float* in_b = (const float*)d_in[2];
    const float* Wqkv = (const float*)d_in[3];
    const float* Wout = (const float*)d_in[4];
    const float* bout = (const float*)d_in[5];
    const float* ln1w = (const float*)d_in[6];
    const float* ln1b = (const float*)d_in[7];
    const float* ln2w = (const float*)d_in[8];
    const float* ln2b = (const float*)d_in[9];
    const float* W1   = (const float*)d_in[10];
    const float* b1   = (const float*)d_in[11];
    const float* W2   = (const float*)d_in[12];
    const float* b2   = (const float*)d_in[13];
    const float* Wg1  = (const float*)d_in[14];
    const float* bg1  = (const float*)d_in[15];
    const float* Wg2  = (const float*)d_in[16];
    const float* bg2  = (const float*)d_in[17];
    const float* lnfw = (const float*)d_in[18];
    const float* lnfb = (const float*)d_in[19];

    const size_t MD = (size_t)Mrows * Dm;
    float* h    = (float*)d_ws;      // (T*B, D)
    float* lnb  = h   + MD;          // LN output / attn output (reused)
    float* y    = lnb + MD;          // proj / FFN output
    float* wide = y   + MD;          // qkv / gate-preact / FFN-mid (M x 1024 max)

    dim3 blk256(256);
    const int ew_blocks = (Mrows * Dm) / 256;

#define GEMM(A0, l0, A1, l1, K1, Wp, Bp, Cp, N, K, RL)                         \
    gemm_bf16_wmma<<<dim3((N) / 64, Mrows / 128), blk256, 0, stream>>>(        \
        A0, l0, A1, l1, K1, Wp, Bp, Cp, N, K, RL)

    // h = x @ in_w^T + in_b ; h += PE
    GEMM(x, Dm, x, Dm, Dm, in_w, in_b, h, Dm, Dm, 0);
    add_pe_k<<<ew_blocks, blk256, 0, stream>>>(h);

    for (int i = 0; i < Ll; ++i) {
        // ---- attention branch ----
        layernorm_k<<<Mrows, blk256, 0, stream>>>(h, ln1w + i * Dm, ln1b + i * Dm, lnb);
        GEMM(lnb, Dm, lnb, Dm, Dm, Wqkv + (size_t)i * 3 * Dm * Dm, nullptr, wide, 3 * Dm, Dm, 0);
        attn_wmma<<<Bb * Hh, blk256, 0, stream>>>(wide, lnb);   // lnb now holds O
        GEMM(lnb, Dm, lnb, Dm, Dm, Wout + (size_t)i * Dm * Dm, bout + i * Dm, y, Dm, Dm, 0);
        // gate1: g = concat(h, y) @ Wg1^T  (split-A GEMM, K=512)
        GEMM(h, Dm, y, Dm, Dm, Wg1 + (size_t)i * 3 * Dm * 2 * Dm, nullptr, wide, 3 * Dm, 2 * Dm, 0);
        gate_k<<<ew_blocks, blk256, 0, stream>>>(h, wide, bg1 + i * Dm);

        // ---- FFN branch ----
        layernorm_k<<<Mrows, blk256, 0, stream>>>(h, ln2w + i * Dm, ln2b + i * Dm, lnb);
        GEMM(lnb, Dm, lnb, Dm, Dm, W1 + (size_t)i * DFFf * Dm, b1 + i * DFFf, wide, DFFf, Dm, 1);
        GEMM(wide, DFFf, wide, DFFf, DFFf, W2 + (size_t)i * Dm * DFFf, b2 + i * Dm, y, Dm, DFFf, 0);
        GEMM(h, Dm, y, Dm, Dm, Wg2 + (size_t)i * 3 * Dm * 2 * Dm, nullptr, wide, 3 * Dm, 2 * Dm, 0);
        gate_k<<<ew_blocks, blk256, 0, stream>>>(h, wide, bg2 + i * Dm);
    }

    layernorm_k<<<Mrows, blk256, 0, stream>>>(h, lnfw, lnfb, (float*)d_out);
#undef GEMM
}
